// GATencoderlayer_49959059587663
// MI455X (gfx1250) — matrix operand order repaired and checked by reading, not compiled, not run on hardware
//
#include <hip/hip_runtime.h>

// ---------------------------------------------------------------------------
// GAT encoder layer for MI455X (gfx1250, wave32).
//   B=8, N=2048, H=1, D=256, in_dim=256.
// Roofline: mandatory HBM traffic ~260MB (adj tril read 67MB, I_A write
// 134MB, out/gate/feat ~60MB) -> ~11us at 23.3TB/s; P@h (17.2 GFLOP) in
// bf16 WMMA 16x16x32 sits far below that. Streaming tensors use NT hints so
// the 192MB L2 keeps W/Hw/h_bf resident.
// Workspace: h_bf16-paired (8.4MB) + gate (16.8MB) + a_src/a_dst (128KB).
// ---------------------------------------------------------------------------

typedef __attribute__((ext_vector_type(2)))  float  v2f;
typedef __attribute__((ext_vector_type(4)))  float  v4f;
typedef __attribute__((ext_vector_type(8)))  float  v8f;
typedef __attribute__((ext_vector_type(2)))  int    v2i;
typedef __attribute__((ext_vector_type(16))) __bf16 v16bf;

#define B_ 8
#define N_ 2048
#define D_ 256

static __device__ __forceinline__ unsigned short f2bf(float x) {
  union { float f; unsigned int u; } c; c.f = x;
  unsigned int r = c.u + 0x7FFFu + ((c.u >> 16) & 1u);   // round-nearest-even
  return (unsigned short)(r >> 16);
}
// fast sigmoid: v_exp_f32 + v_rcp_f32 (no precise-div chain)
static __device__ __forceinline__ float sigmoidf_(float x) {
  return __builtin_amdgcn_rcpf(1.0f + __expf(-x));
}

// ---------------------------------------------------------------------------
// Kernel 1: per 16-row tile. 8 waves; wave w computes 16x16 d-blocks {w,w+8}
// for BOTH h=feat@W and gatePre=feat@Hw^T via V_WMMA_F32_16X16X4_F32 (K=256).
// Branchless inner loop: B operand address = bbase[k*kstride], selected per
// job outside the K loop.
// f32 WMMA layouts (ISA 7.12.2): lane L, m=n=L&15, hl=L>>4.
//   A(16x4): a[v]=A[m][k+2hl+v]  B(4x16): b[v]=B[k+2hl+v][n]  C: c[r]=C[r+8hl][n]
// h is written to hbf in PAIRED-ROW bf16 layout:
//   uint32 word [b][p][d] = { lo16: row 2p , hi16: row 2p+1 } at column d
// C-fragment rows 2q,2q+1 live in the same lane -> pack in-register and emit
// 4 coalesced b32 stores per job.
// ---------------------------------------------------------------------------
__global__ __launch_bounds__(256)
void gat_prep(const float* __restrict__ feat, const float* __restrict__ Wm,
              const float* __restrict__ Ws,  const float* __restrict__ Hw,
              const float* __restrict__ Hb,
              unsigned int* __restrict__ hpair, float* __restrict__ gate,
              float* __restrict__ asrc, float* __restrict__ adst)
{
  __shared__ __align__(16) float ldsFeat[16][D_ + 2];  // +2 pad: bank-safe b64
  __shared__ float accS[16], accD[16];

  const int tile = blockIdx.x;
  const int b    = tile / (N_ / 16);
  const int i0   = (tile % (N_ / 16)) * 16;
  const int t    = threadIdx.x;
  const int wave = t >> 5, lane = t & 31;
  const int m    = lane & 15, hl = lane >> 4;

  if (t < 16) { accS[t] = 0.0f; accD[t] = 0.0f; }

  // stage feat tile (row-padded LDS): row = t>>4, 16 threads x 4 float4 each
  {
    const int row = t >> 4;
    const float* src = feat + ((size_t)b * N_ + i0 + row) * D_;
    #pragma unroll
    for (int q = 0; q < 4; ++q) {
      const int c = ((t & 15) + q * 16) * 4;
      *(v4f*)&ldsFeat[row][c] = *(const v4f*)(src + c);
    }
  }
  __syncthreads();

  for (int job = 0; job < 4; ++job) {
    const int  db  = (job & 1) ? wave + 8 : wave;
    const int  d   = db * 16 + m;               // output column for this lane
    const bool isH = (job < 2);
    // branchless B addressing: W is [k][d] (stride D_), Hw^T is [d][k] (stride 1)
    const float* bbase = isH ? (Wm + d) : (Hw + (size_t)d * D_);
    const size_t kstr  = isH ? (size_t)D_ : 1;

    v8f acc = {};
    #pragma unroll 8
    for (int k = 0; k < D_; k += 4) {
      const int k0 = k + 2 * hl;
      v2f a, bf;
      a[0]  = ldsFeat[m][k0];
      a[1]  = ldsFeat[m][k0 + 1];
      bf[0] = bbase[(size_t)k0 * kstr];
      bf[1] = bbase[(size_t)(k0 + 1) * kstr];
      acc = __builtin_amdgcn_wmma_f32_16x16x4_f32(
              false, a, false, bf, (short)0, acc, false, false);
    }
    if (isH) {
      const float wss = Ws[d], wsd = Ws[D_ + d];
      float ps = 0.0f, pd = 0.0f;
      #pragma unroll
      for (int q = 0; q < 4; ++q) {             // rows 8hl+2q, 8hl+2q+1
        const int   row0 = 8 * hl + 2 * q;
        const float h0 = acc[2 * q], h1 = acc[2 * q + 1];
        hpair[((size_t)b * (N_ >> 1) + ((i0 + row0) >> 1)) * D_ + d] =
            (unsigned int)f2bf(h0) | ((unsigned int)f2bf(h1) << 16);
        atomicAdd(&accS[row0],     h0 * wss);   // ds_add_f32
        atomicAdd(&accS[row0 + 1], h1 * wss);
        atomicAdd(&accD[row0],     h0 * wsd);
        atomicAdd(&accD[row0 + 1], h1 * wsd);
      }
      (void)ps; (void)pd;
    } else {
      const float hb = Hb[d];
      #pragma unroll
      for (int r = 0; r < 8; ++r) {
        const int row = r + 8 * hl;
        gate[((size_t)b * N_ + i0 + row) * D_ + d] = sigmoidf_(acc[r] + hb);
      }
    }
  }
  __syncthreads();
  if (t < 16) {
    asrc[(size_t)b * N_ + i0 + t] = accS[t];
    adst[(size_t)b * N_ + i0 + t] = accD[t];
  }
}

// ---------------------------------------------------------------------------
// Kernel 2: per 16-row tile, stream 32-col chunks up to the diagonal.
// P tile (16x32 bf16) double-buffered in LDS -> single barrier per chunk.
// B fragments come straight from global h (paired-row layout, L2-resident).
// bf16 WMMA layouts, lane L, n=L&15, hl=L>>4:
//   A(16x32): v<4 -> K={2v,2v+1}+8hl ; v>=4 -> K={2v+8,2v+9}+8hl
//   B(32x16): same K pair as A at column n  == one uint of paired h
// ---------------------------------------------------------------------------
__global__ __launch_bounds__(256)
void gat_attn(const float* __restrict__ feat, const int* __restrict__ adj,
              const float* __restrict__ bvec, const float* __restrict__ bs,
              const unsigned int* __restrict__ hpair,
              const float* __restrict__ gate,
              const float* __restrict__ asrc, const float* __restrict__ adst,
              float* __restrict__ out, float* __restrict__ IA)
{
  __shared__ unsigned int ldsP[2][16 * 17];   // 16x32 bf16 pairs, padded
  __shared__ float rs[16];
  __shared__ float sA[16];

  const int tile = blockIdx.x;
  const int b    = tile / (N_ / 16);
  const int i0   = (tile % (N_ / 16)) * 16;
  const int t    = threadIdx.x;
  const int wave = t >> 5, lane = t & 31;
  const int n    = lane & 15, hl = lane >> 4;
  const int d0   = wave * 16, d1 = d0 + 128;  // two d-blocks per wave

  const int pm = t >> 4;            // P-row (0..15), fixed per thread
  const int pk = (t & 15) * 2;      // P-col pair base (0..30)
  const float bsc = bs[0];

  if (t < 16) { rs[t] = 0.0f; sA[t] = asrc[(size_t)b * N_ + i0 + t]; }
  __syncthreads();

  const int    i_row   = i0 + pm;
  const size_t rowBase = ((size_t)b * N_ + i_row) * N_;     // adj / IA row
  const size_t hBase   = (size_t)b * (N_ >> 1) * D_;        // paired h base
  const float  myAs    = sA[pm];
  float rsum_local = 0.0f;
  v8f acc0 = {}, acc1 = {};

  const int jcmax = (i0 + 15) >> 5;
  for (int jc = 0; jc <= jcmax; ++jc) {
    const int j0  = jc << 5;
    const int buf = jc & 1;

    // ---- P pair + I_A pair for (i_row, cols j0+pk, j0+pk+1) ----
    {
      const int j = j0 + pk;
      const v2i ad  = __builtin_nontemporal_load((const v2i*)(adj + rowBase + j));
      const v2f ad2 = *(const v2f*)(adst + (size_t)b * N_ + j);
      float ia[2], p[2];
      #pragma unroll
      for (int e = 0; e < 2; ++e) {
        const int je = j + e;
        float attn = 0.0f;
        if (je < i_row) attn = sigmoidf_(myAs + ad2[e] + bsc);
        ia[e] = (je == i_row ? 1.0f : 0.0f) - attn;
        const bool allowed = (je <= i_row) && (ad[e] == 1);
        p[e] = allowed ? __expf(ia[e] - 1.0f) : 0.0f;   // fixed max ref = 1
      }
      v2f iav; iav[0] = ia[0]; iav[1] = ia[1];
      __builtin_nontemporal_store(iav, (v2f*)(IA + rowBase + j));
      ldsP[buf][pm * 17 + (pk >> 1)] =
          (unsigned int)f2bf(p[0]) | ((unsigned int)f2bf(p[1]) << 16);
      rsum_local += p[0] + p[1];
    }
    if (jc < jcmax)   // prefetch next adj chunk (global_prefetch_b8)
      __builtin_prefetch(adj + rowBase + j0 + 32 + pk, 0, 1);
    __syncthreads();  // P(buf) visible; prev chunk's wmma used the other buf

    // ---- WMMA: A = P tile (LDS), B = paired h from global (L2-hot) ----
    union Frag { v16bf v; unsigned int w[8]; };
    Frag A, B0, B1;
    #pragma unroll
    for (int v = 0; v < 8; ++v) {
      const int Ka = ((v < 4) ? (2 * v) : (2 * v + 8)) + 8 * hl;
      A.w[v] = ldsP[buf][n * 17 + (Ka >> 1)];
      const int Kb = j0 + Ka;                         // even
      const size_t pr = hBase + (size_t)(Kb >> 1) * D_;
      B0.w[v] = hpair[pr + d0 + n];
      B1.w[v] = hpair[pr + d1 + n];
    }
    acc0 = __builtin_amdgcn_wmma_f32_16x16x32_bf16(
             false, A.v, false, B0.v, (short)0, acc0, false, false);
    acc1 = __builtin_amdgcn_wmma_f32_16x16x32_bf16(
             false, A.v, false, B1.v, (short)0, acc1, false, false);
  }

  atomicAdd(&rs[pm], rsum_local);

  // zero-fill I_A above the covered chunks (diag already inside chunks)
  {
    const int jhi = (jcmax + 1) << 5;
    const v4f z4 = {0.0f, 0.0f, 0.0f, 0.0f};
    for (int c = jhi + (t & 15) * 4; c < N_; c += 64)
      __builtin_nontemporal_store(z4, (v4f*)(IA + rowBase + c));
  }
  __syncthreads();

  // epilogue: normalize (v_rcp), +bias, elu, gate blend -> out (streamed, NT)
  #pragma unroll
  for (int r = 0; r < 8; ++r) {
    const int row = r + 8 * hl;
    float rsum = rs[row];
    rsum = (rsum > 1e-30f) ? rsum : 1e-30f;     // all-masked-row guard
    const float rinv = __builtin_amdgcn_rcpf(rsum);
    const size_t o = ((size_t)b * N_ + i0 + row) * D_;
    #pragma unroll
    for (int s = 0; s < 2; ++s) {
      const int   d = (s ? d1 : d0) + n;
      const float v = (s ? acc1[r] : acc0[r]) * rinv + bvec[d];
      const float e = (v > 0.0f) ? v : (__expf(v) - 1.0f);
      const float g  = __builtin_nontemporal_load(gate + o + d);
      const float fi = __builtin_nontemporal_load(feat + o + d);
      __builtin_nontemporal_store(g * e + (1.0f - g) * fi, out + o + d);
    }
  }
}

// ---------------------------------------------------------------------------
extern "C" void kernel_launch(void* const* d_in, const int* in_sizes, int n_in,
                              void* d_out, int out_size, void* d_ws, size_t ws_size,
                              hipStream_t stream) {
  (void)in_sizes; (void)n_in; (void)out_size; (void)ws_size;
  const float* feat = (const float*)d_in[0];
  const int*   adj  = (const int*)  d_in[1];
  const float* Wm   = (const float*)d_in[2];
  const float* bias = (const float*)d_in[3];
  const float* Ws   = (const float*)d_in[4];
  const float* bs   = (const float*)d_in[5];
  const float* Hw   = (const float*)d_in[6];
  const float* Hb   = (const float*)d_in[7];

  float* out = (float*)d_out;
  float* IA  = out + (size_t)B_ * N_ * D_;     // tuple: (out, I_A_raw)

  unsigned int* hpair = (unsigned int*)d_ws;                    //  8.4 MB
  float*        gate  = (float*)(hpair + (size_t)B_ * (N_ >> 1) * D_); // 16.8 MB
  float*        asrc  = gate + (size_t)B_ * N_ * D_;            // 64 KB
  float*        adst  = asrc + (size_t)B_ * N_;                 // 64 KB

  const dim3 grid(B_ * N_ / 16), blk(256);
  gat_prep<<<grid, blk, 0, stream>>>(feat, Wm, Ws, Hw, Hb, hpair, gate, asrc, adst);
  gat_attn<<<grid, blk, 0, stream>>>(feat, adj, bias, bs, hpair, gate,
                                     asrc, adst, out, IA);
}
